// CTGRUCell_31550829756936
// MI455X (gfx1250) — compile-verified, matrix operand order checked by reading, not run on hardware
//
#include <hip/hip_runtime.h>

// ---------------------------------------------------------------------------
// CT-GRU cell for MI455X (gfx1250, wave32, WMMA + async global->LDS DMA).
// Dominant cost: three 1024x1024x4096 GEMMs -> v_wmma_f32_16x16x32_bf16,
// tiles staged with global_load_async_to_lds_b128 (ASYNCcnt-tracked DMA).
// ---------------------------------------------------------------------------

typedef __attribute__((ext_vector_type(16))) __bf16 v16bf;
typedef __attribute__((ext_vector_type(8)))  float  v8f;

constexpr int Bsz = 1024;            // batch
constexpr int IN  = 512;             // input size
constexpr int N_U = 512;             // units
constexpr int M_S = 8;               // scales
constexpr int FAN = IN + N_U;        // 1024 (K of the GEMMs)
constexpr int NM  = N_U * M_S;       // 4096 (N of the GEMMs)

__device__ __forceinline__ unsigned short f2bf(float f) {
    unsigned int u = __float_as_uint(f);
    u = (u + 0x7fffu + ((u >> 16) & 1u)) >> 16;   // round-to-nearest-even
    return (unsigned short)u;
}

// Async DMA: copy 16B per lane from global to LDS (no VGPR data path).
// The LDS operand is derived from the real shared pointer (ptrtoint) so the
// shared object escapes -> the "memory" clobber makes the compiler treat the
// DMA as a writer of LDS (keeps the ds_load fragment reads live and ordered).
// Low 32 bits of a generic LDS-aperture address == LDS byte offset (ISA 10.2).
__device__ __forceinline__ void async_copy16(const void* ldsPtr, const void* gptr) {
    const unsigned off = (unsigned)(size_t)ldsPtr;
    asm volatile("global_load_async_to_lds_b128 %0, %1, off"
                 :: "v"(off), "v"(gptr) : "memory");
}
#define WAIT_ASYNC(n) asm volatile("s_wait_asynccnt " #n ::: "memory")

// ---------------------------------------------------------------------------
// Transpose + f32->bf16 convert:  W (1024 x 4096, row major) -> Wt (4096 x 1024)
// ---------------------------------------------------------------------------
__global__ __launch_bounds__(256) void k_transpose_bf16(
    const float* __restrict__ W, unsigned short* __restrict__ Wt)
{
    __shared__ float tile[32][33];
    const int tx = threadIdx.x, ty = threadIdx.y;
    const int x = blockIdx.x * 32 + tx;           // col of W (N dim)
    const int y = blockIdx.y * 32 + ty;           // row of W (K dim)
    #pragma unroll
    for (int j = 0; j < 32; j += 8)
        tile[ty + j][tx] = W[(size_t)(y + j) * NM + x];
    __syncthreads();
    const int ox = blockIdx.y * 32 + tx;          // col of Wt (K dim)
    const int oy = blockIdx.x * 32 + ty;          // row of Wt (N dim)
    #pragma unroll
    for (int j = 0; j < 32; j += 8)
        Wt[(size_t)(oy + j) * FAN + ox] = f2bf(tile[tx][ty + j]);
}

// ---------------------------------------------------------------------------
// fused = [inputs | sum_m h]  as bf16  (1024 x 1024)
// ---------------------------------------------------------------------------
__global__ __launch_bounds__(256) void k_fuse1(
    const float* __restrict__ inp, const float* __restrict__ st,
    unsigned short* __restrict__ fused)
{
    const int tid = blockIdx.x * 256 + threadIdx.x;   // < 1024*1024
    const int b = tid >> 10, j = tid & 1023;
    float v;
    if (j < IN) {
        v = inp[b * IN + j];
    } else {
        const float4* p = (const float4*)(st + (size_t)b * NM + (j - IN) * 8);
        float4 a = p[0], c = p[1];
        v = a.x + a.y + a.z + a.w + c.x + c.y + c.z + c.w;
    }
    fused[tid] = f2bf(v);
}

// ---------------------------------------------------------------------------
// fused_r = [inputs | sum_m r*h] as bf16
// ---------------------------------------------------------------------------
__global__ __launch_bounds__(256) void k_fuse2(
    const float* __restrict__ inp, const float* __restrict__ st,
    const float* __restrict__ r, unsigned short* __restrict__ fused)
{
    const int tid = blockIdx.x * 256 + threadIdx.x;
    const int b = tid >> 10, j = tid & 1023;
    float v;
    if (j < IN) {
        v = inp[b * IN + j];
    } else {
        const size_t base = (size_t)b * NM + (j - IN) * 8;
        const float4* ph = (const float4*)(st + base);
        const float4* pr = (const float4*)(r + base);
        float4 h0 = ph[0], h1 = ph[1], r0 = pr[0], r1 = pr[1];
        v = h0.x * r0.x + h0.y * r0.y + h0.z * r0.z + h0.w * r0.w
          + h1.x * r1.x + h1.y * r1.y + h1.z * r1.z + h1.w * r1.w;
    }
    fused[tid] = f2bf(v);
}

// ---------------------------------------------------------------------------
// WMMA GEMM: C(1024x4096) = A(1024x1024,bf16) * Bt(4096x1024,bf16)^T
// Block: 256 threads = 8 waves (2x4), block tile 128x128, k-step 32.
// Tiles staged by async global->LDS DMA, double-buffered; each wave issues
// 4 async b128 loads per k-step and overlaps them with 8 WMMAs.
// mode 0: out = sigmoid(acc + bias)                      (gate z / r)
// mode 1: out = (1-z)*h + z*alpha[m]*tanh(acc + bias)    (new_state)
// ---------------------------------------------------------------------------
__global__ __launch_bounds__(256) void k_gemm_wmma(
    const unsigned short* __restrict__ A,
    const unsigned short* __restrict__ Bt,
    const float* __restrict__ bias,
    float* __restrict__ out,
    int mode,
    const float* __restrict__ zbuf,
    const float* __restrict__ hstate)
{
    // Double-buffered LDS: per buffer two 128x32 bf16 tiles, row stride padded
    // to 20 dwords (80B) -> 16B-aligned, bank-conflict-free ds_load_b128.
    constexpr int TILE_Q = 128 * 20;          // dwords per tile
    constexpr int BUF_Q  = 2 * TILE_Q;        // dwords per buffer (A + B)
    __shared__ unsigned int lds[2 * BUF_Q];   // 40 KB

    const int tid = threadIdx.x;
    const int bM = blockIdx.y;                // 0..7   (batch rows / 128)
    const int bN = blockIdx.x;                // 0..31  (out cols  / 128)

    // --- staging assignment: thread t DMAs 32B of one tile row ---
    const int rStage = tid >> 1;              // 0..127
    const int cHalf  = tid & 1;               // which 32B half of the 64B row
    const unsigned short* gA = A  + (size_t)(bM * 128 + rStage) * FAN + cHalf * 16;
    const unsigned short* gB = Bt + (size_t)(bN * 128 + rStage) * FAN + cHalf * 16;
    const unsigned ldsQ = rStage * 20 + cHalf * 8;   // dword offset in tile

    // --- wave layout ---
    const int lane  = tid & 31;
    const int wv    = tid >> 5;
    const int wrow  = wv >> 2;                // 0..1  -> 64 batch rows each
    const int wcol  = wv & 3;                 // 0..3  -> 32 out cols each
    const int l15   = lane & 15;
    const int lhalf = lane >> 4;              // selects K-base 0 / 8 in fragment

    v8f acc[4][2];
    {
        v8f zero = {};
        #pragma unroll
        for (int tm = 0; tm < 4; tm++)
            #pragma unroll
            for (int tn = 0; tn < 2; tn++)
                acc[tm][tn] = zero;
    }

    union Frag { v16bf v; uint4 q[2]; };

    // issue async DMA for one k-step into buffer `buf`
    auto issue = [&](int buf, int k0) {
        unsigned int* dstA = &lds[buf * BUF_Q + ldsQ];
        unsigned int* dstB = dstA + TILE_Q;
        async_copy16(dstA,     gA + k0);
        async_copy16(dstA + 4, gA + k0 + 8);
        async_copy16(dstB,     gB + k0);
        async_copy16(dstB + 4, gB + k0 + 8);
    };

    constexpr int STEPS = FAN / 32;           // 32
    issue(0, 0);

    #pragma unroll
    for (int i = 0; i < STEPS; ++i) {
        const int k0 = i * 32;
        const int cur = i & 1;
        if (i + 1 < STEPS) {
            issue(cur ^ 1, k0 + 32);          // DMA next tile (overlaps WMMAs)
            WAIT_ASYNC(4);                    // current buffer's 4 loads landed
        } else {
            WAIT_ASYNC(0);
        }
        __syncthreads();                      // all waves' DMA visible

        const unsigned int* As = lds + cur * BUF_Q;
        const unsigned int* Bs = As + TILE_Q;

        // B fragments (column n in lane l15; K packed 0..7/16..23 | 8..15/24..31)
        Frag bf[2];
        #pragma unroll
        for (int tn = 0; tn < 2; tn++) {
            const int r   = wcol * 32 + tn * 16 + l15;
            const int off = r * 20 + lhalf * 4;
            bf[tn].q[0] = *(const uint4*)&Bs[off];
            bf[tn].q[1] = *(const uint4*)&Bs[off + 8];
        }
        #pragma unroll
        for (int tm = 0; tm < 4; tm++) {
            Frag af;
            const int r   = wrow * 64 + tm * 16 + l15;
            const int off = r * 20 + lhalf * 4;
            af.q[0] = *(const uint4*)&As[off];
            af.q[1] = *(const uint4*)&As[off + 8];
            #pragma unroll
            for (int tn = 0; tn < 2; tn++) {
                acc[tm][tn] = __builtin_amdgcn_wmma_f32_16x16x32_bf16(
                    false, af.v, false, bf[tn].v,
                    (short)0, acc[tm][tn], false, false);
            }
        }
        __syncthreads();                      // guard WAR on buffer reuse
    }

    // --- epilogue ---
    // C/D layout: lane 0-15 -> N=lane, M=vgpr; lane 16-31 -> N=lane-16, M=vgpr+8
    const int rowBase = bM * 128 + wrow * 64;
    const int colBase = bN * 128 + wcol * 32;
    #pragma unroll
    for (int tn = 0; tn < 2; tn++) {
        const int col = colBase + tn * 16 + l15;
        const float bv = bias[col];
        float al = 0.0f;
        if (mode == 1) {
            // alpha[m] = 10^(-m/2) / sum_m 10^(-m/2);  sum = 1.462329048
            const int m = col & 7;
            al = __expf(-1.1512925465f * (float)m) * 0.6838412f;
        }
        #pragma unroll
        for (int tm = 0; tm < 4; tm++) {
            #pragma unroll
            for (int v = 0; v < 8; v++) {
                const int row = rowBase + tm * 16 + v + lhalf * 8;
                const size_t idx = (size_t)row * NM + col;
                const float x = acc[tm][tn][v] + bv;
                if (mode == 0) {
                    out[idx] = 1.0f / (1.0f + __expf(-x));
                } else {
                    const float hh = tanhf(x);
                    const float zz = zbuf[idx];
                    const float h  = hstate[idx];
                    out[idx] = (1.0f - zz) * h + zz * al * hh;
                }
            }
        }
    }
}

// ---------------------------------------------------------------------------
// output[b,n] = sum_m new_state[b, n*8+m]
// ---------------------------------------------------------------------------
__global__ __launch_bounds__(256) void k_reduceM(
    const float* __restrict__ ns, float* __restrict__ outp)
{
    const int tid = blockIdx.x * 256 + threadIdx.x;   // < 1024*512
    const float4* p = (const float4*)(ns + (size_t)tid * 8);
    float4 a = p[0], c = p[1];
    outp[tid] = a.x + a.y + a.z + a.w + c.x + c.y + c.z + c.w;
}

// ---------------------------------------------------------------------------
extern "C" void kernel_launch(void* const* d_in, const int* in_sizes, int n_in,
                              void* d_out, int out_size, void* d_ws, size_t ws_size,
                              hipStream_t stream)
{
    const float* inputs = (const float*)d_in[0];   // (1024, 512)
    const float* state  = (const float*)d_in[1];   // (1024, 4096)
    const float* W_z    = (const float*)d_in[2];   // (1024, 4096)
    const float* b_z    = (const float*)d_in[3];   // (4096,)
    const float* W_r    = (const float*)d_in[4];
    const float* b_r    = (const float*)d_in[5];
    const float* W_h    = (const float*)d_in[6];
    const float* b_h    = (const float*)d_in[7];

    float* out      = (float*)d_out;                       // (1024, 512)
    float* newstate = out + (size_t)Bsz * N_U;             // (1024, 4096)

    // workspace carve-up (60 MB total)
    char* ws = (char*)d_ws;
    unsigned short* Wzt    = (unsigned short*)ws;  ws += (size_t)NM * FAN * 2; // 8 MB
    unsigned short* Wrt    = (unsigned short*)ws;  ws += (size_t)NM * FAN * 2; // 8 MB
    unsigned short* Wht    = (unsigned short*)ws;  ws += (size_t)NM * FAN * 2; // 8 MB
    unsigned short* fused  = (unsigned short*)ws;  ws += (size_t)Bsz * FAN * 2; // 2 MB
    unsigned short* fusedr = (unsigned short*)ws;  ws += (size_t)Bsz * FAN * 2; // 2 MB
    float*          zbuf   = (float*)ws;           ws += (size_t)Bsz * NM * 4;  // 16 MB
    float*          rbuf   = (float*)ws;                                        // 16 MB

    // 1) transpose+convert weights to bf16 (N x K)
    dim3 tb(32, 8), tg(NM / 32, FAN / 32);
    k_transpose_bf16<<<tg, tb, 0, stream>>>(W_z, Wzt);
    k_transpose_bf16<<<tg, tb, 0, stream>>>(W_r, Wrt);
    k_transpose_bf16<<<tg, tb, 0, stream>>>(W_h, Wht);

    // 2) fused = [inputs | h_sum]
    k_fuse1<<<(Bsz * FAN) / 256, 256, 0, stream>>>(inputs, state, fused);

    // 3) gates z, r
    dim3 gg(NM / 128, Bsz / 128);   // (32, 8)
    k_gemm_wmma<<<gg, 256, 0, stream>>>(fused, Wzt, b_z, zbuf, 0, nullptr, nullptr);
    k_gemm_wmma<<<gg, 256, 0, stream>>>(fused, Wrt, b_r, rbuf, 0, nullptr, nullptr);

    // 4) fused_r = [inputs | sum_m r*h]
    k_fuse2<<<(Bsz * FAN) / 256, 256, 0, stream>>>(inputs, state, rbuf, fusedr);

    // 5) candidate GEMM + state update, writes new_state straight into d_out
    k_gemm_wmma<<<gg, 256, 0, stream>>>(fusedr, Wht, b_h, newstate, 1, zbuf, state);

    // 6) output = sum over M
    k_reduceM<<<(Bsz * N_U) / 256, 256, 0, stream>>>(newstate, out);
}
